// SelectiveSSM_22016002359646
// MI455X (gfx1250) — matrix operand order repaired and checked by reading, not compiled, not run on hardware
//
#include <hip/hip_runtime.h>
#include <math.h>

#define S_LEN   2048
#define DMODEL  768
#define DINNER  1536
#define DSTATE  16
#define NPROJ   33   // 1 (dt) + 16 (B) + 16 (C)

typedef __attribute__((ext_vector_type(16))) __bf16 v16bf;
typedef __attribute__((ext_vector_type(8)))  float  v8f;

// fp32 -> bf16 round-to-nearest-even
static __device__ __forceinline__ unsigned short f2bf(float f) {
    unsigned int u = __float_as_uint(f);
    unsigned int r = (u + 0x7FFFu + ((u >> 16) & 1u)) >> 16;
    return (unsigned short)r;
}

static __device__ __forceinline__ float silu(float v) {
    return v / (1.0f + __expf(-v));
}

// ---------------- conversion kernels ----------------
__global__ void k_cvt(const float* __restrict__ src, unsigned short* __restrict__ dst, int n) {
    int i = blockIdx.x * blockDim.x + threadIdx.x;
    if (i < n) dst[i] = f2bf(src[i]);
}

// src [rows][cols] fp32 row-major -> dst [cols][rows] bf16 (N-major weights for B fragments)
__global__ void k_cvt_t(const float* __restrict__ src, unsigned short* __restrict__ dst,
                        int rows, int cols) {
    int i = blockIdx.x * blockDim.x + threadIdx.x;
    if (i < rows * cols) {
        int r = i / cols, c = i % cols;
        dst[(size_t)c * rows + r] = f2bf(src[i]);
    }
}

// ---------------- bf16 WMMA GEMM: C[M][N] = A[M][K] @ BT[N][K]^T ----------------
// Each wave computes a 16x64 C tile: one A fragment is reused across 4 WMMAs
// (4 independent accumulators -> XDL ILP + 4x less A traffic). 8 waves/block
// cover 128 M rows. tri!=0: lower-triangular A (K dim == M dim); K loop runs
// only to the row-tile end (upper triangle stored as zeros, so rounding the
// bound up to a multiple of 32 is safe).
__global__ __launch_bounds__(256) void k_gemm(const unsigned short* __restrict__ A,
                                              const unsigned short* __restrict__ BT,
                                              float* __restrict__ C,
                                              int M, int N, int K, int tri) {
    const int wave = threadIdx.x >> 5;
    const int lane = threadIdx.x & 31;
    const int l = lane & 15;     // M row (A) / N col (B) within 16-wide tile
    const int h = lane >> 4;     // lane half selects K sub-block
    const int m0 = blockIdx.y * 128 + wave * 16;
    const int n0 = blockIdx.x * 64;

    const unsigned short* arow = A + (size_t)(m0 + l) * K;
    const unsigned short* brow0 = BT + (size_t)(n0 + 0  + l) * K;
    const unsigned short* brow1 = BT + (size_t)(n0 + 16 + l) * K;
    const unsigned short* brow2 = BT + (size_t)(n0 + 32 + l) * K;
    const unsigned short* brow3 = BT + (size_t)(n0 + 48 + l) * K;

    int kmax = K;
    if (tri) { int kl = (m0 + 16 + 31) & ~31; kmax = kl < K ? kl : K; }

    v8f acc0 = {}, acc1 = {}, acc2 = {}, acc3 = {};
    union Frag { v16bf v; uint4 q[2]; };
    Frag fa, fb0, fb1, fb2, fb3;

    for (int k0 = 0; k0 < kmax; k0 += 32) {
        // A 16x32 bf16 fragment: lane half h holds contiguous K = k0+16h .. +15 per row
        fa.q[0] = *reinterpret_cast<const uint4*>(arow + k0 + 16 * h);
        fa.q[1] = *reinterpret_cast<const uint4*>(arow + k0 + 16 * h + 8);
        // B 32x16 bf16 fragments: lane half h holds contiguous K = k0+16h .. +15 per column
        const int bo = k0 + 16 * h;
        fb0.q[0] = *reinterpret_cast<const uint4*>(brow0 + bo);
        fb0.q[1] = *reinterpret_cast<const uint4*>(brow0 + bo + 8);
        fb1.q[0] = *reinterpret_cast<const uint4*>(brow1 + bo);
        fb1.q[1] = *reinterpret_cast<const uint4*>(brow1 + bo + 8);
        fb2.q[0] = *reinterpret_cast<const uint4*>(brow2 + bo);
        fb2.q[1] = *reinterpret_cast<const uint4*>(brow2 + bo + 8);
        fb3.q[0] = *reinterpret_cast<const uint4*>(brow3 + bo);
        fb3.q[1] = *reinterpret_cast<const uint4*>(brow3 + bo + 8);
        acc0 = __builtin_amdgcn_wmma_f32_16x16x32_bf16(false, fa.v, false, fb0.v,
                                                       (short)0, acc0, false, false);
        acc1 = __builtin_amdgcn_wmma_f32_16x16x32_bf16(false, fa.v, false, fb1.v,
                                                       (short)0, acc1, false, false);
        acc2 = __builtin_amdgcn_wmma_f32_16x16x32_bf16(false, fa.v, false, fb2.v,
                                                       (short)0, acc2, false, false);
        acc3 = __builtin_amdgcn_wmma_f32_16x16x32_bf16(false, fa.v, false, fb3.v,
                                                       (short)0, acc3, false, false);
    }

    // C/D layout: VGPR j -> M = j + 8*half, N = lane%16
#pragma unroll
    for (int j = 0; j < 8; ++j) {
        float* crow = C + (size_t)(m0 + j + 8 * h) * N + n0 + l;
        crow[0]  = acc0[j];
        crow[16] = acc1[j];
        crow[32] = acc2[j];
        crow[48] = acc3[j];
    }
}

// ---------------- depthwise conv (width 4, pad 1/2) + SiLU ----------------
__global__ void k_conv_silu(const float* __restrict__ xz, const float* __restrict__ cw,
                            const float* __restrict__ cb, float* __restrict__ xb, int n) {
    int i = blockIdx.x * blockDim.x + threadIdx.x;
    if (i >= n) return;
    int s = i / DINNER, d = i % DINNER;
    float acc = cb[d];
#pragma unroll
    for (int k = 0; k < 4; ++k) {
        int sp = s + k - 1;
        if (sp >= 0 && sp < S_LEN)
            acc += xz[(size_t)sp * (2 * DINNER) + d] * cw[k * DINNER + d];
    }
    xb[i] = silu(acc);
}

// ---------------- proj = xb @ W_xproj (N=33) -> dt (softplus), B, C ----------------
__global__ void k_proj(const float* __restrict__ xb, const float* __restrict__ W,
                       float* __restrict__ dt, float* __restrict__ Bm,
                       float* __restrict__ Cm) {
    int s = blockIdx.x, j = threadIdx.x;
    if (j >= NPROJ) return;
    const float* row = xb + (size_t)s * DINNER;
    float acc = 0.f;
    for (int d = 0; d < DINNER; ++d) acc += row[d] * W[(size_t)d * NPROJ + j];
    if (j == 0) {
        dt[s] = (acc > 20.f) ? acc : log1pf(__expf(acc));
    } else if (j <= DSTATE) {
        Bm[(size_t)s * DSTATE + (j - 1)] = acc;
    } else {
        Cm[(size_t)s * DSTATE + (j - 1 - DSTATE)] = acc;
    }
}

// ---------------- u[t,d] = xb[t,d] * mean_n exp(-exp(A_log[d,n]) * dt[t]) -> uT bf16 ----------------
__global__ void k_u(const float* __restrict__ xb, const float* __restrict__ dt,
                    const float* __restrict__ A_log, unsigned short* __restrict__ uT, int n) {
    int i = blockIdx.x * blockDim.x + threadIdx.x;
    if (i >= n) return;
    int s = i / DINNER, d = i % DINNER;
    float t = dt[s], w = 0.f;
#pragma unroll
    for (int nn = 0; nn < DSTATE; ++nn)
        w += __expf(-__expf(A_log[d * DSTATE + nn]) * t);
    w *= (1.f / DSTATE);
    uT[(size_t)d * S_LEN + s] = f2bf(xb[i] * w);
}

// ---------------- K[s,t] = (t<=s) ? dt[t] * <B_t, C_s> : 0  (bf16) ----------------
__global__ void k_buildK(const float* __restrict__ dt, const float* __restrict__ Bm,
                         const float* __restrict__ Cm, unsigned short* __restrict__ Kb, int n) {
    int i = blockIdx.x * blockDim.x + threadIdx.x;
    if (i >= n) return;
    int s = i / S_LEN, t = i % S_LEN;
    float v = 0.f;
    if (t <= s) {
        const float* b = Bm + (size_t)t * DSTATE;
        const float* c = Cm + (size_t)s * DSTATE;
        float acc = 0.f;
#pragma unroll
        for (int nn = 0; nn < DSTATE; ++nn) acc += b[nn] * c[nn];
        v = dt[t] * acc;
    }
    Kb[i] = f2bf(v);
}

// ---------------- y2 = yscan * silu(z) + xb * D  (bf16 for final GEMM) ----------------
__global__ void k_combine(const float* __restrict__ ys, const float* __restrict__ xz,
                          const float* __restrict__ xb, const float* __restrict__ Dp,
                          unsigned short* __restrict__ y2, int n) {
    int i = blockIdx.x * blockDim.x + threadIdx.x;
    if (i >= n) return;
    int s = i / DINNER, d = i % DINNER;
    float z = xz[(size_t)s * (2 * DINNER) + DINNER + d];
    y2[i] = f2bf(ys[i] * silu(z) + xb[i] * Dp[d]);
}

// ---------------- residual add + LayerNorm (eps = 1e-3, biased var) ----------------
__global__ __launch_bounds__(256) void k_ln(const float* __restrict__ mm,
                                            const float* __restrict__ x,
                                            const float* __restrict__ gamma,
                                            const float* __restrict__ beta,
                                            float* __restrict__ out) {
    __shared__ float rs[256], rq[256];
    int s = blockIdx.x, t = threadIdx.x;
    float h[3];
    float sum = 0.f, sq = 0.f;
#pragma unroll
    for (int i = 0; i < 3; ++i) {
        int d = t + i * 256;
        float v = mm[(size_t)s * DMODEL + d] + x[(size_t)s * DMODEL + d];
        h[i] = v; sum += v; sq += v * v;
    }
    rs[t] = sum; rq[t] = sq;
    __syncthreads();
    for (int o = 128; o > 0; o >>= 1) {
        if (t < o) { rs[t] += rs[t + o]; rq[t] += rq[t + o]; }
        __syncthreads();
    }
    float mu   = rs[0] * (1.f / DMODEL);
    float var  = rq[0] * (1.f / DMODEL) - mu * mu;
    float rstd = rsqrtf(var + 1e-3f);
#pragma unroll
    for (int i = 0; i < 3; ++i) {
        int d = t + i * 256;
        out[(size_t)s * DMODEL + d] = gamma[d] * (h[i] - mu) * rstd + beta[d];
    }
}

extern "C" void kernel_launch(void* const* d_in, const int* in_sizes, int n_in,
                              void* d_out, int out_size, void* d_ws, size_t ws_size,
                              hipStream_t stream) {
    (void)in_sizes; (void)n_in; (void)out_size; (void)ws_size;
    const float* x      = (const float*)d_in[0];
    const float* W_in   = (const float*)d_in[1];
    const float* conv_w = (const float*)d_in[2];
    const float* conv_b = (const float*)d_in[3];
    const float* W_xp   = (const float*)d_in[4];
    const float* A_log  = (const float*)d_in[5];
    const float* D_par  = (const float*)d_in[6];
    const float* W_out  = (const float*)d_in[7];
    const float* gamma  = (const float*)d_in[8];
    const float* beta   = (const float*)d_in[9];
    float* out = (float*)d_out;

    char* p = (char*)d_ws;
    auto carve = [&](size_t bytes) -> char* {
        char* r = p; p += (bytes + 255) & ~(size_t)255; return r;
    };
    float* xz    = (float*)carve(sizeof(float) * S_LEN * 2 * DINNER);   // 25.2 MB
    float* xbp   = (float*)carve(sizeof(float) * S_LEN * DINNER);       // 12.6 MB
    float* yscan = (float*)carve(sizeof(float) * S_LEN * DINNER);       // 12.6 MB
    float* mmO   = (float*)carve(sizeof(float) * S_LEN * DMODEL);       //  6.3 MB
    float* dtA   = (float*)carve(sizeof(float) * S_LEN);
    float* BmA   = (float*)carve(sizeof(float) * S_LEN * DSTATE);
    float* CmA   = (float*)carve(sizeof(float) * S_LEN * DSTATE);
    unsigned short* xb16  = (unsigned short*)carve(2u * S_LEN * DMODEL);
    unsigned short* WinT  = (unsigned short*)carve(2u * 2 * DINNER * DMODEL);
    unsigned short* uT    = (unsigned short*)carve(2u * DINNER * S_LEN);
    unsigned short* K16   = (unsigned short*)carve(2u * S_LEN * S_LEN);
    unsigned short* y216  = (unsigned short*)carve(2u * S_LEN * DINNER);
    unsigned short* WoutT = (unsigned short*)carve(2u * DMODEL * DINNER);

    const int T = 256;
    auto blocks = [](int n, int t) { return (n + t - 1) / t; };

    // 1) xz = x @ W_in  (bf16 WMMA, fp32 accumulate)
    k_cvt  <<<blocks(S_LEN * DMODEL, T), T, 0, stream>>>(x, xb16, S_LEN * DMODEL);
    k_cvt_t<<<blocks(DMODEL * 2 * DINNER, T), T, 0, stream>>>(W_in, WinT, DMODEL, 2 * DINNER);
    k_gemm <<<dim3((2 * DINNER) / 64, S_LEN / 128), 256, 0, stream>>>(
        xb16, WinT, xz, S_LEN, 2 * DINNER, DMODEL, 0);

    // 2) conv + SiLU, 3) proj -> dt/B/C
    k_conv_silu<<<blocks(S_LEN * DINNER, T), T, 0, stream>>>(xz, conv_w, conv_b, xbp,
                                                             S_LEN * DINNER);
    k_proj<<<S_LEN, 64, 0, stream>>>(xbp, W_xp, dtA, BmA, CmA);

    // 4) scan as triangular GEMM: y = K @ u
    k_u     <<<blocks(S_LEN * DINNER, T), T, 0, stream>>>(xbp, dtA, A_log, uT, S_LEN * DINNER);
    k_buildK<<<blocks(S_LEN * S_LEN, T), T, 0, stream>>>(dtA, BmA, CmA, K16, S_LEN * S_LEN);
    k_gemm  <<<dim3(DINNER / 64, S_LEN / 128), 256, 0, stream>>>(
        K16, uT, yscan, S_LEN, DINNER, S_LEN, 1);

    // 5) gate + skip, 6) y @ W_out
    k_combine<<<blocks(S_LEN * DINNER, T), T, 0, stream>>>(yscan, xz, xbp, D_par, y216,
                                                           S_LEN * DINNER);
    k_cvt_t<<<blocks(DINNER * DMODEL, T), T, 0, stream>>>(W_out, WoutT, DINNER, DMODEL);
    k_gemm <<<dim3(DMODEL / 64, S_LEN / 128), 256, 0, stream>>>(
        y216, WoutT, mmO, S_LEN, DMODEL, DINNER, 0);

    // 7) residual + LayerNorm
    k_ln<<<S_LEN, 256, 0, stream>>>(mmO, x, gamma, beta, out);
}
